// DWConv2d_BN_71150428225676
// MI455X (gfx1250) — compile-verified, matrix operand order checked by reading, not run on hardware
//
#include <hip/hip_runtime.h>
#include <hip/hip_fp16.h>

#define BATCH 8
#define CIN   192
#define HH    56
#define WW    56
#define HWSZ  (HH * WW)      // 3136
#define OUTC  384
#define NOFF  18

typedef __attribute__((ext_vector_type(16))) _Float16 v16h;
typedef __attribute__((ext_vector_type(8)))  float    v8f;

union Frag16 { v16h h; unsigned int u[8]; };

// ---------------------------------------------------------------------------
// Kernel 1: offset generation. Fused depthwise 3x3 (pad=1, groups=C) +
// pointwise C->18 + clip[-1,1]. One thread per (b,h,w); 18 fp32 accumulators.
// og_pw/og_dw indices are thread-uniform -> scalarized (s_load) by compiler.
// ---------------------------------------------------------------------------
__global__ void __launch_bounds__(256) offsets_kernel(
    const float* __restrict__ x, const float* __restrict__ og_dw,
    const float* __restrict__ og_pw, float* __restrict__ off) {
  int idx = blockIdx.x * blockDim.x + threadIdx.x;
  if (idx >= BATCH * HWSZ) return;
  int b = idx / HWSZ;
  int hw = idx - b * HWSZ;
  int h = hw / WW, w = hw - (hw / WW) * WW;

  float acc[NOFF];
#pragma unroll
  for (int j = 0; j < NOFF; ++j) acc[j] = 0.0f;

  const float* xb = x + (size_t)b * CIN * HWSZ;
  for (int c = 0; c < CIN; ++c) {
    const float* xc = xb + (size_t)c * HWSZ;
    const float* wd = og_dw + c * 9;
    float dwv = 0.0f;
#pragma unroll
    for (int t = 0; t < 9; ++t) {
      int yy = h + (t / 3) - 1;
      int xx = w + (t % 3) - 1;
      float v = (yy >= 0 && yy < HH && xx >= 0 && xx < WW) ? xc[yy * WW + xx] : 0.0f;
      dwv = fmaf(v, wd[t], dwv);
    }
#pragma unroll
    for (int j = 0; j < NOFF; ++j)
      acc[j] = fmaf(dwv, og_pw[j * CIN + c], acc[j]);
  }
  float* ob = off + (size_t)b * NOFF * HWSZ;
#pragma unroll
  for (int j = 0; j < NOFF; ++j)
    ob[j * HWSZ + hw] = fminf(1.0f, fmaxf(-1.0f, acc[j]));
}

// ---------------------------------------------------------------------------
// Kernel 2: pw_w (O,C) fp32 -> fp16 (keeps K contiguous for WMMA B frags)
// ---------------------------------------------------------------------------
__global__ void __launch_bounds__(256) cvt_w_kernel(
    const float* __restrict__ pw_w, __half* __restrict__ pw16) {
  int i = blockIdx.x * blockDim.x + threadIdx.x;
  if (i < OUTC * CIN) pw16[i] = __float2half(pw_w[i]);
}

__device__ __forceinline__ float samp(const float* __restrict__ xc, int yy, int xx) {
  return (yy >= 0 && yy < HH && xx >= 0 && xx < WW) ? xc[yy * WW + xx] : 0.0f;
}

// ---------------------------------------------------------------------------
// Kernel 3: deformable bilinear sampling + depthwise reduce over 9 taps.
// Block = 192 threads (one per channel, 6 waves); offsets for the block's
// (b,h,w) staged once in LDS. Output stored TRANSPOSED as f16: (B*HW, C)
// with C contiguous -> A-matrix of the WMMA GEMM is K-contiguous.
// ---------------------------------------------------------------------------
__global__ void __launch_bounds__(192) deform_dw_kernel(
    const float* __restrict__ x, const float* __restrict__ off,
    const float* __restrict__ dcn_w, __half* __restrict__ ydT) {
  __shared__ float soff[NOFF];
  int bhw = blockIdx.x;
  int b = bhw / HWSZ;
  int hw = bhw - b * HWSZ;
  int h = hw / WW, w = hw - (hw / WW) * WW;

  if (threadIdx.x < NOFF)
    soff[threadIdx.x] = off[((size_t)b * NOFF + threadIdx.x) * HWSZ + hw];
  __syncthreads();

  int c = threadIdx.x;
  const float* xc = x + ((size_t)b * CIN + c) * HWSZ;
  const float* wd = dcn_w + c * 9;

  float acc = 0.0f;
#pragma unroll
  for (int k = 0; k < 9; ++k) {
    float ys = (float)(h + (k / 3) - 1) + soff[2 * k + 0];
    float xs = (float)(w + (k % 3) - 1) + soff[2 * k + 1];
    float y0f = floorf(ys), x0f = floorf(xs);
    float ty = ys - y0f, tx = xs - x0f;
    int y0 = (int)y0f, x0 = (int)x0f;
    float v = samp(xc, y0,     x0    ) * (1.0f - ty) * (1.0f - tx)
            + samp(xc, y0,     x0 + 1) * (1.0f - ty) * tx
            + samp(xc, y0 + 1, x0    ) * ty * (1.0f - tx)
            + samp(xc, y0 + 1, x0 + 1) * ty * tx;
    acc = fmaf(v, wd[k], acc);
  }
  ydT[(size_t)bhw * CIN + c] = __float2half(acc);
}

// ---------------------------------------------------------------------------
// Kernel 4: pointwise 192->384 GEMM via v_wmma_f32_16x16x32_f16 + hardswish.
// M = B*HW = 25088, N = 384, K = 192 (6 K-chunks, fp32 accumulation).
// Each wave computes a 16(M) x 32(N) tile: the A fragment of each K-chunk is
// reused across two N tiles (12 WMMAs per wave, A vmem traffic halved).
// Fragment layout per ISA 7.12.2 (16-bit A, 16x32): lane row = lane&15,
// hi = lane>=16, VGPR v holds K = (v&4)<<2 | (v&3)<<1 | hi<<3 and K+1.
// A/B buffers are K-contiguous f16 -> fragments lower to global_load_b128.
// Batch index is tile-uniform (3136 % 16 == 0) -> no per-row division, and
// the 8 output floats per lane are contiguous -> global_store_b128 x2.
// All guards are wave-uniform so EXEC is all-ones at every WMMA.
// ---------------------------------------------------------------------------
__global__ void __launch_bounds__(256) pw_wmma_kernel(
    const __half* __restrict__ ydT, const __half* __restrict__ pw16,
    float* __restrict__ out) {
  const int MT  = (BATCH * HWSZ) / 16;  // 1568
  const int NP  = OUTC / 32;            // 12 N-tile pairs
  int wave = threadIdx.x >> 5;
  int lane = threadIdx.x & 31;
  int tile = blockIdx.x * 8 + wave;     // grid sized exactly: 18816 / 8 waves
  if (tile >= MT * NP) return;          // wave-uniform (never taken)

  int mtile = tile / NP;
  int npair = tile - mtile * NP;
  int row = lane & 15;
  int hi  = lane >> 4;

  const __half* arow  = ydT  + (size_t)(mtile * 16 + row) * CIN;
  const __half* brow0 = pw16 + (size_t)(npair * 32 + row) * CIN;
  const __half* brow1 = brow0 + (size_t)16 * CIN;

  v8f acc0 = {};
  v8f acc1 = {};
#pragma unroll
  for (int kk = 0; kk < CIN; kk += 32) {
    Frag16 a, b0, b1;
#pragma unroll
    for (int v = 0; v < 8; ++v) {
      int kb = kk + (((v & 4) << 2) | ((v & 3) << 1) | (hi << 3));
      a.u[v]  = *(const unsigned int*)(arow  + kb);
      b0.u[v] = *(const unsigned int*)(brow0 + kb);
      b1.u[v] = *(const unsigned int*)(brow1 + kb);
    }
    acc0 = __builtin_amdgcn_wmma_f32_16x16x32_f16(
        false, a.h, false, b0.h, (short)0, acc0, false, false);
    acc1 = __builtin_amdgcn_wmma_f32_16x16x32_f16(
        false, a.h, false, b1.h, (short)0, acc1, false, false);
  }

  // D layout: VGPR r -> M = r + 8*hi, N = lane&15. Hardswish, store NCHW.
  // Batch / hw base are tile-uniform: computed once, no per-row division.
  int mbase  = mtile * 16;
  int b      = mbase / HWSZ;                 // uniform -> scalar
  int hwbase = mbase - b * HWSZ + hi * 8;    // + per-half-wave offset
  int ncol0  = npair * 32 + row;
  float* o0 = out + ((size_t)b * OUTC + ncol0) * HWSZ + hwbase;
  float* o1 = o0 + (size_t)16 * HWSZ;
#pragma unroll
  for (int r = 0; r < 8; ++r) {
    float y0 = acc0[r];
    o0[r] = y0 * fminf(fmaxf(y0 + 3.0f, 0.0f), 6.0f) * (1.0f / 6.0f);
    float y1 = acc1[r];
    o1[r] = y1 * fminf(fmaxf(y1 + 3.0f, 0.0f), 6.0f) * (1.0f / 6.0f);
  }
}

// ---------------------------------------------------------------------------
extern "C" void kernel_launch(void* const* d_in, const int* in_sizes, int n_in,
                              void* d_out, int out_size, void* d_ws, size_t ws_size,
                              hipStream_t stream) {
  const float* x     = (const float*)d_in[0];
  const float* og_dw = (const float*)d_in[1];
  const float* og_pw = (const float*)d_in[2];
  const float* dcn_w = (const float*)d_in[3];
  const float* pw_w  = (const float*)d_in[4];
  float* out = (float*)d_out;

  // Workspace partition (needs ~12.2 MB):
  //   [0, 2MB)      off  : (B, 18, HW) fp32   = 1.81 MB
  //   [2MB, 12MB)   ydT  : (B*HW, C)  fp16    = 9.63 MB
  //   [12MB, ...)   pw16 : (O, C)     fp16    = 147 KB
  char* ws = (char*)d_ws;
  float*  off  = (float*)(ws);
  __half* ydT  = (__half*)(ws + (size_t)2  * 1024 * 1024);
  __half* pw16 = (__half*)(ws + (size_t)12 * 1024 * 1024);

  offsets_kernel<<<(BATCH * HWSZ + 255) / 256, 256, 0, stream>>>(x, og_dw, og_pw, off);
  cvt_w_kernel<<<(OUTC * CIN + 255) / 256, 256, 0, stream>>>(pw_w, pw16);
  deform_dw_kernel<<<BATCH * HWSZ, 192, 0, stream>>>(x, off, dcn_w, ydT);

  const int wave_tiles = ((BATCH * HWSZ) / 16) * (OUTC / 32);  // 18816
  pw_wmma_kernel<<<wave_tiles / 8, 256, 0, stream>>>(ydT, pw16, out);
}
